// TemporalGNN_14800457302608
// MI455X (gfx1250) — compile-verified, matrix operand order checked by reading, not run on hardware
//
#include <hip/hip_runtime.h>
#include <hip/hip_bf16.h>

typedef __attribute__((ext_vector_type(16))) _Float16 v16h;
typedef __attribute__((ext_vector_type(8)))  float    v8f;
typedef __attribute__((ext_vector_type(4)))  int      v4i_t;

#define S_TOT 192      // B*T snapshots
#define NN    1000     // nodes
#define FF    64       // features
#define EE    16000    // edges
#define KDIM  576      // 9 bases * 64
#define NCOL  128      // z (64) ++ h (64)
#define NRB   63       // ceil(1000/16) row blocks
#define ASTR  584      // padded LDS stride for A tile (halves)

// CDNA5 async global->LDS copy path (ASYNCcnt-tracked), with safe fallback.
#if defined(__has_builtin)
#if __has_builtin(__builtin_amdgcn_global_load_async_to_lds_b128) && \
    __has_builtin(__builtin_amdgcn_s_wait_asynccnt)
#define HAVE_ASYNC_LDS 1
#endif
#endif

#ifdef HAVE_ASYNC_LDS
#define GLB_AS __attribute__((address_space(1)))
#define LDS_AS __attribute__((address_space(3)))
__device__ __forceinline__ void async_cp_b128(const void* gsrc, void* ldst) {
    __builtin_amdgcn_global_load_async_to_lds_b128(
        (GLB_AS v4i_t*)gsrc, (LDS_AS v4i_t*)ldst, 0, 0);
}
__device__ __forceinline__ void async_wait0() {
    __builtin_amdgcn_s_wait_asynccnt(0);
}
#else
__device__ __forceinline__ void async_cp_b128(const void* gsrc, void* ldst) {
    *reinterpret_cast<float4*>(ldst) = *reinterpret_cast<const float4*>(gsrc);
}
__device__ __forceinline__ void async_wait0() {}
#endif

// ---------------- utility ----------------
__global__ void k_zero_f32(float* p, long long n) {
    long long i = (long long)blockIdx.x * blockDim.x + threadIdx.x;
    if (i < n) p[i] = 0.0f;
}

// ---------------- degrees / norms ----------------
__global__ void k_degrees(const int* ei, float* deg_out, float* deg_in) {
    int e = blockIdx.x * blockDim.x + threadIdx.x;
    if (e < EE) {
        atomicAdd(&deg_out[ei[e]], 1.0f);        // row = src
        atomicAdd(&deg_in[ei[EE + e]], 1.0f);    // col = dst
    }
}
__global__ void k_edge_norm(const int* ei, const float* deg_out, const float* deg_in,
                            float* norm_out, float* norm_in) {
    int e = blockIdx.x * blockDim.x + threadIdx.x;
    if (e < EE) {
        int r = ei[e];
        norm_out[e] = 1.0f / deg_out[r];
        norm_in[e]  = 1.0f / deg_in[r];
    }
}

// ---------------- weight packing: Bt[col][576] fp16 ----------------
// col<64 -> z gate, col>=64 -> h gate; kd = b*64 + fin
// b==0: W[0,0]+W[1,0] ; b in 1..4: W[0,b] ; b in 5..8: W[1,b-4]
// W layout [2][5][128][64]; only fin rows [0:64) matter (h0 == 0).
__global__ void k_pack_w(const float* Wz, const float* Wh, _Float16* Bt) {
    int i = blockIdx.x * blockDim.x + threadIdx.x;
    if (i >= NCOL * KDIM) return;
    int col = i / KDIM, kd = i % KDIM;
    int b = kd >> 6, fin = kd & 63, fout = col & 63;
    const float* W = (col < 64) ? Wz : Wh;
    float v;
    if (b == 0)
        v = W[(0 * 5 + 0) * 8192 + fin * 64 + fout] + W[(1 * 5 + 0) * 8192 + fin * 64 + fout];
    else if (b <= 4)
        v = W[(0 * 5 + b) * 8192 + fin * 64 + fout];
    else
        v = W[(1 * 5 + (b - 4)) * 8192 + fin * 64 + fout];
    Bt[(size_t)col * KDIM + kd] = (_Float16)v;
}

// ---------------- pack x into basis slot 0 ----------------
__global__ void k_pack_x(const float* x, _Float16* Apack) {
    long long i = (long long)blockIdx.x * blockDim.x + threadIdx.x;
    if (i >= (long long)S_TOT * NN * FF) return;
    long long sn = i >> 6;          // s*NN + n
    int f = (int)(i & 63);
    Apack[sn * KDIM + f] = (_Float16)x[i];
}

// ---------------- propagation scatter: scratch[a] += Apack[b][slot]*norm ----------------
__global__ void k_prop(const _Float16* Apack, int srcSlot,
                       const int* aIdx, const int* bIdx, const float* norm,
                       float* scratch) {
    int s = blockIdx.y;
    int eLocal = threadIdx.x >> 6;         // 4 edges per block
    int f = threadIdx.x & 63;
    int e = blockIdx.x * 4 + eLocal;
    if (e >= EE) return;
    int a = aIdx[e], b = bIdx[e];
    float w = norm[e];
    float v = (float)Apack[((size_t)s * NN + b) * KDIM + srcSlot * 64 + f] * w;
    atomicAdd(&scratch[((size_t)s * NN + a) * FF + f], v);
}

// ---------------- Chebyshev combine: T_new = scale*P - T_{k-2}, store fp16 ----------------
__global__ void k_combine(_Float16* Apack, const float* scratch,
                          int outSlot, float scale, int subSlot) {
    long long i = (long long)blockIdx.x * blockDim.x + threadIdx.x;
    if (i >= (long long)S_TOT * NN * FF) return;
    long long sn = i >> 6;
    int f = (int)(i & 63);
    float v = scale * scratch[i];
    if (subSlot >= 0) v -= (float)Apack[sn * KDIM + subSlot * 64 + f];
    Apack[sn * KDIM + outSlot * 64 + f] = (_Float16)v;
}

// ---------------- fused WMMA cell: GEMM + gates + ReLU + LayerNorm ----------------
__global__ __launch_bounds__(256)
void k_gemm_cell(const _Float16* __restrict__ Apack, const _Float16* __restrict__ Bt,
                 const float* __restrict__ b_z, const float* __restrict__ b_h,
                 const float* __restrict__ ln_g, const float* __restrict__ ln_b,
                 float* __restrict__ h_final) {
    __shared__ _Float16 A_lds[16 * ASTR];     // 16 rows x 576 (padded)
    __shared__ _Float16 B_lds[NCOL * 64];     // 128 cols x 64-K chunk
    __shared__ float    tile[16 * NCOL];      // f32 accum tile (z ++ h)
    __shared__ float    pre[16 * 64];         // post-gate pre-LN values
    __shared__ float    mu_s[16], rs_s[16];

    const int s   = blockIdx.y;
    const int n0  = blockIdx.x * 16;
    const int tid = threadIdx.x;
    const int lane = tid & 31, wv = tid >> 5;
    const int laneLo = lane & 15, hi = lane >> 4;

    // cooperative A-tile load: 16 x 576 halves = 1152 b128 transfers (async -> LDS)
    for (int i = tid; i < 16 * 72; i += 256) {
        int r = i / 72, j = i % 72;
        int n = n0 + r;
        _Float16* dst = &A_lds[r * ASTR + j * 8];
        if (n < NN)
            async_cp_b128(Apack + ((size_t)s * NN + n) * KDIM + j * 8, dst);
        else
            *reinterpret_cast<float4*>(dst) = make_float4(0.f, 0.f, 0.f, 0.f);
    }

    v8f acc = {0.f, 0.f, 0.f, 0.f, 0.f, 0.f, 0.f, 0.f};

    for (int kk = 0; kk < KDIM; kk += 64) {
        __syncthreads();   // all reads of previous B chunk complete
        // async-load B chunk: 128 cols x 64 K-halves = 1024 b128 transfers
        for (int i = tid; i < 1024; i += 256) {
            int c = i >> 3, j = i & 7;
            async_cp_b128(Bt + (size_t)c * KDIM + kk + j * 8, &B_lds[c * 64 + j * 8]);
        }
        // prefetch next chunk toward L2 while this one lands in LDS
        if (kk + 64 < KDIM)
            __builtin_prefetch(Bt + (size_t)(tid >> 1) * KDIM + kk + 64, 0, 0);
        async_wait0();     // ASYNCcnt == 0: LDS writes visible
        __syncthreads();
#pragma unroll
        for (int k2 = 0; k2 < 64; k2 += 32) {
            // A fragment: row = laneLo, halves {K+0..7, K+16..23}, hi lanes +8
            const _Float16* pa = &A_lds[laneLo * ASTR + kk + k2 + hi * 8];
            v16h a;
            reinterpret_cast<float4*>(&a)[0] = *reinterpret_cast<const float4*>(pa);
            reinterpret_cast<float4*>(&a)[1] = *reinterpret_cast<const float4*>(pa + 16);
            // B fragment: col = wv*16 + laneLo, same K packing (Bt is pre-transposed)
            const _Float16* pb = &B_lds[(wv * 16 + laneLo) * 64 + k2 + hi * 8];
            v16h b;
            reinterpret_cast<float4*>(&b)[0] = *reinterpret_cast<const float4*>(pb);
            reinterpret_cast<float4*>(&b)[1] = *reinterpret_cast<const float4*>(pb + 16);
            acc = __builtin_amdgcn_wmma_f32_16x16x32_f16(
                      false, a, false, b, (short)0, acc, false, false);
        }
    }
    __syncthreads();

    // D layout: lane n = laneLo (col in tile), vgpr g -> row m = hi*8 + g
#pragma unroll
    for (int g = 0; g < 8; ++g)
        tile[(hi * 8 + g) * NCOL + wv * 16 + laneLo] = acc[g];
    __syncthreads();

    // gate combine: h = relu((1 - sigmoid(z + bz)) * tanh(ht + bh))
    for (int i = tid; i < 16 * 64; i += 256) {
        int m = i >> 6, f = i & 63;
        float z  = tile[m * NCOL + f] + b_z[f];
        float ht = tile[m * NCOL + 64 + f] + b_h[f];
        float sg = 1.0f / (1.0f + __expf(-z));
        float v  = (1.0f - sg) * tanhf(ht);
        pre[i] = v > 0.0f ? v : 0.0f;
    }
    __syncthreads();

    if (tid < 16) {
        float mu = 0.0f;
        for (int f = 0; f < 64; ++f) mu += pre[tid * 64 + f];
        mu *= (1.0f / 64.0f);
        float var = 0.0f;
        for (int f = 0; f < 64; ++f) {
            float d = pre[tid * 64 + f] - mu;
            var += d * d;
        }
        var *= (1.0f / 64.0f);
        mu_s[tid] = mu;
        rs_s[tid] = rsqrtf(var + 1e-5f);
    }
    __syncthreads();

    for (int i = tid; i < 16 * 64; i += 256) {
        int m = i >> 6, f = i & 63;
        int n = n0 + m;
        if (n < NN)
            h_final[((size_t)s * NN + n) * FF + f] =
                (pre[i] - mu_s[m]) * rs_s[m] * ln_g[f] + ln_b[f];
    }
}

// ---------------- final linear ----------------
__global__ void k_out_init(const float* lin_b, float* out) {
    int i = blockIdx.x * blockDim.x + threadIdx.x;
    if (i < 160) out[i] = lin_b[i % 10];
}
__global__ void k_final_linear(const float* __restrict__ hfin,
                               const float* __restrict__ lin_w,
                               float* __restrict__ out) {
    __shared__ float red[256];
    int b = blockIdx.z, c = blockIdx.y;
    long long base = (long long)blockIdx.x * 1024;
    const float* hp = hfin + (size_t)b * 768000;
    const float* wp = lin_w + (size_t)c * 768000;
    float sum = 0.0f;
#pragma unroll
    for (int j = 0; j < 4; ++j) {
        long long i = base + threadIdx.x + j * 256;
        sum += hp[i] * wp[i];
    }
    red[threadIdx.x] = sum;
    __syncthreads();
    for (int w = 128; w > 0; w >>= 1) {
        if (threadIdx.x < w) red[threadIdx.x] += red[threadIdx.x + w];
        __syncthreads();
    }
    if (threadIdx.x == 0) atomicAdd(&out[b * 10 + c], red[0]);
}

// ---------------- host launch ----------------
extern "C" void kernel_launch(void* const* d_in, const int* in_sizes, int n_in,
                              void* d_out, int out_size, void* d_ws, size_t ws_size,
                              hipStream_t stream) {
    const float* x     = (const float*)d_in[0];
    const int*   ei    = (const int*)  d_in[1];
    const float* Wz    = (const float*)d_in[2];
    const float* bz    = (const float*)d_in[3];
    // d_in[4], d_in[5] = W_r, b_r : provably unused (r * h0 == 0)
    const float* Wh    = (const float*)d_in[6];
    const float* bh    = (const float*)d_in[7];
    const float* lng   = (const float*)d_in[8];
    const float* lnb   = (const float*)d_in[9];
    const float* linw  = (const float*)d_in[10];
    const float* linb  = (const float*)d_in[11];
    float* out = (float*)d_out;

    char* ws = (char*)d_ws;
    float*     deg_out  = (float*)(ws + 0);
    float*     deg_in   = (float*)(ws + 4096);
    float*     norm_out = (float*)(ws + 8192);
    float*     norm_in  = (float*)(ws + 8192 + 65536);
    _Float16*  Bt       = (_Float16*)(ws + 139264);
    _Float16*  Apack    = (_Float16*)(ws + 286720);                 // 221,184,000 B
    float*     scratch  = (float*)(ws + 286720 + 221184000ULL);     //  49,152,000 B
    float*     h_final  = (float*)(ws + 286720 + 221184000ULL + 49152000ULL);

    const int* rowIdx = ei;        // edge_index[0]
    const int* colIdx = ei + EE;   // edge_index[1]
    const long long SNF = (long long)S_TOT * NN * FF;
    const int snfBlocks = (int)((SNF + 255) / 256);

    // 1) degrees + per-edge norms
    k_zero_f32<<<8, 256, 0, stream>>>(deg_out, 2048);   // deg_out|deg_in contiguous
    k_degrees<<<(EE + 255) / 256, 256, 0, stream>>>(ei, deg_out, deg_in);
    k_edge_norm<<<(EE + 255) / 256, 256, 0, stream>>>(ei, deg_out, deg_in, norm_out, norm_in);

    // 2) packing
    k_pack_w<<<(NCOL * KDIM + 255) / 256, 256, 0, stream>>>(Wz, Wh, Bt);
    k_pack_x<<<snfBlocks, 256, 0, stream>>>(x, Apack);

    // 3) Chebyshev recurrence, both directions (slots: 0=Tx0, 1..4=out, 5..8=in)
    // out-dir: dst=col, src=row, coeff=1/deg_out[row]  | in-dir: dst=row, src=col, coeff=1/deg_in[row]
    struct Step { int srcSlot, outSlot, subSlot; float scale; };
    const Step steps[4] = { {0, 0, -1, 1.0f}, {1, 1, 0, 2.0f}, {2, 2, 1, 2.0f}, {3, 3, 2, 2.0f} };
    dim3 pgrid(EE / 4, S_TOT, 1);
    for (int dir = 0; dir < 2; ++dir) {
        const int* aIdx = (dir == 0) ? colIdx : rowIdx;
        const int* bIdx = (dir == 0) ? rowIdx : colIdx;
        const float* nrm = (dir == 0) ? norm_out : norm_in;
        int offs = dir * 4;  // slot offset for this direction
        for (int k = 0; k < 4; ++k) {
            Step st = steps[k];
            int srcSlot = (st.srcSlot == 0) ? 0 : st.srcSlot + offs;
            int outSlot = st.outSlot + 1 + offs;
            int subSlot = (st.subSlot < 0) ? -1 : ((st.subSlot == 0) ? 0 : st.subSlot + offs);
            k_zero_f32<<<snfBlocks, 256, 0, stream>>>(scratch, SNF);
            k_prop<<<pgrid, 256, 0, stream>>>(Apack, srcSlot, aIdx, bIdx, nrm, scratch);
            k_combine<<<snfBlocks, 256, 0, stream>>>(Apack, scratch, outSlot, st.scale, subSlot);
        }
    }

    // 4) fused WMMA GEMM + gates + ReLU + LayerNorm
    dim3 ggrid(NRB, S_TOT, 1);
    k_gemm_cell<<<ggrid, 256, 0, stream>>>(Apack, Bt, bz, bh, lng, lnb, h_final);

    // 5) final linear projection
    k_out_init<<<1, 256, 0, stream>>>(linb, out);
    dim3 fgrid(750, 10, 16);
    k_final_linear<<<fgrid, 256, 0, stream>>>(h_final, linw, out);

    (void)in_sizes; (void)n_in; (void)out_size; (void)ws_size;
}
